// LSTMResidualNet_29618094473949
// MI455X (gfx1250) — matrix operand order repaired and checked by reading, not compiled, not run on hardware
//
#include <hip/hip_runtime.h>

typedef __attribute__((ext_vector_type(16))) _Float16 v16h;
typedef __attribute__((ext_vector_type(8)))  _Float16 v8h;
typedef __attribute__((ext_vector_type(8)))  float    v8f;

static constexpr int Bb = 512;
static constexpr int Tt = 256;

__device__ __forceinline__ float sigf(float x) { return 1.0f / (1.0f + __expf(-x)); }

// ---------------------------------------------------------------- f32 -> f16
__global__ void cvt_f32_f16(const float* __restrict__ in, _Float16* __restrict__ out, int n) {
  for (int i = blockIdx.x * blockDim.x + threadIdx.x; i < n; i += gridDim.x * blockDim.x)
    out[i] = (_Float16)in[i];
}

// ------------------------------------------------------------------ gate GEMM
// XG[m, n'] = sum_k A[m,k] * w_ih[n_orig,k] + b_ih[n_orig] + b_hh[n_orig]
// Column permutation: n' = u*4 + gate, n_orig = gate*H + u  (gate order i,f,g,o)
// One wave owns a 16-row M tile; whole (permuted, f16) w_ih lives in LDS.
template<int DIN, int N4H>
__global__ void gate_gemm(const _Float16* __restrict__ A,   // [M, DIN] f16
                          const float*    __restrict__ w_ih,// [N4H, DIN] f32
                          const float*    __restrict__ b_ih,
                          const float*    __restrict__ b_hh,
                          _Float16*       __restrict__ xg,  // [M, N4H] f16 (permuted cols)
                          int M)
{
  constexpr int NT = N4H / 16;
  constexpr int H  = N4H / 4;
  constexpr int KF = DIN / 32;
  extern __shared__ char smem[];
  _Float16* Wl = (_Float16*)smem;                               // [N4H][DIN]
  float*    bl = (float*)(smem + (size_t)N4H * DIN * 2);        // [N4H]

  const int tid  = threadIdx.x;
  const int lane = tid & 31;
  const int wave = tid >> 5;
  const int mlo  = lane & 15;
  const int mhi  = lane >> 4;

  for (int idx = tid; idx < N4H * DIN; idx += blockDim.x) {
    int n = idx / DIN, k = idx - n * DIN;
    int u = n >> 2, g = n & 3;
    Wl[idx] = (_Float16)w_ih[(size_t)(g * H + u) * DIN + k];
  }
  for (int n = tid; n < N4H; n += blockDim.x) {
    int u = n >> 2, g = n & 3;
    bl[n] = b_ih[g * H + u] + b_hh[g * H + u];
  }
  __syncthreads();

  const int m0 = blockIdx.x * 128 + wave * 16;
  if (m0 >= M) return;

  // A fragments (16-bit A 16x32 layout): lane m=mlo, chunks k0+mhi*8 and k0+16+mhi*8
  v16h af[KF];
  #pragma unroll
  for (int kf = 0; kf < KF; ++kf) {
    const _Float16* p = A + (size_t)(m0 + mlo) * DIN + kf * 32 + mhi * 8;
    *(v8h*)&af[kf]       = *(const v8h*)p;
    *((v8h*)&af[kf] + 1) = *(const v8h*)(p + 16);
  }

  for (int tn = 0; tn < NT; ++tn) {
    const int n0 = tn * 16;
    const float bv = bl[n0 + mlo];
    v8f acc;
    #pragma unroll
    for (int r = 0; r < 8; ++r) acc[r] = bv;
    #pragma unroll
    for (int kf = 0; kf < KF; ++kf) {
      // B fragment (16-bit B 32x16): lane n=mlo, chunks k0+mhi*8 and k0+16+mhi*8
      v16h bf;
      const _Float16* p = Wl + (size_t)(n0 + mlo) * DIN + kf * 32 + mhi * 8;
      *(v8h*)&bf       = *(const v8h*)p;
      *((v8h*)&bf + 1) = *(const v8h*)(p + 16);
      acc = __builtin_amdgcn_wmma_f32_16x16x32_f16(false, af[kf], false, bf,
                                                   (short)0, acc, false, false);
    }
    #pragma unroll
    for (int r = 0; r < 8; ++r) {
      int m = m0 + mhi * 8 + r;                 // C layout: M = mhi*8 + r, N = mlo
      xg[(size_t)m * N4H + n0 + mlo] = (_Float16)acc[r];
    }
  }
}

// ------------------------------------------------------------------ LSTM scan
// One workgroup (8 waves) owns a 16-row batch tile for all T steps.
// w_hh (f16, gate-permuted, K padded to KP) resident in LDS; h (f16) in LDS;
// c stays in registers of the gate==0 lanes. LN fused per timestep.
// The next timestep's xg tile [16, N4H] is DMA'd into LDS with
// GLOBAL_LOAD_ASYNC_TO_LDS_B128, overlapped with the current step's WMMA work.
template<int H, int N4H, int TPW>
__global__ void lstm_scan(const _Float16* __restrict__ xg,     // [B*T, N4H] permuted
                          const float*    __restrict__ w_hh,   // [N4H, H] original
                          const float*    __restrict__ gamma,
                          const float*    __restrict__ beta,
                          _Float16*       __restrict__ seq_out,// [B*T, H] LN'd
                          int T)
{
  constexpr int KP = (H + 31) & ~31;
  constexpr int NT = N4H / 16;
  extern __shared__ char smem[];
  _Float16* Wl  = (_Float16*)smem;                  // [N4H][KP]
  _Float16* hl  = Wl + (size_t)N4H * KP;            // [16][KP]
  _Float16* xgl = hl + 16 * KP;                     // [16][N4H] async-staged tile

  const int tid  = threadIdx.x;
  const int lane = tid & 31;
  const int wave = tid >> 5;
  const int mlo  = lane & 15;
  const int mhi  = lane >> 4;
  const int gate = lane & 3;
  const int b0   = blockIdx.x * 16;

  // wave-relative LDS byte offset of xgl (dynamic smem starts after static LDS)
  const unsigned xgl_lds = __builtin_amdgcn_groupstaticsize() +
                           (unsigned)(((size_t)N4H * KP + 16 * KP) * 2);

  for (int idx = tid; idx < N4H * KP; idx += blockDim.x) {
    int n = idx / KP, k = idx - n * KP;
    int u = n >> 2, g = n & 3;
    Wl[idx] = (_Float16)((k < H) ? w_hh[(size_t)(g * H + u) * H + k] : 0.0f);
  }
  for (int idx = tid; idx < 16 * KP; idx += blockDim.x) hl[idx] = (_Float16)0.0f;

  // ---- async preload of the t=0 xg tile (B128 per lane, contiguous per row)
  {
    const int NOPS = 2 * N4H;                       // 16*N4H*2 bytes / 16
    for (int idx = tid; idx < NOPS; idx += 256) {
      const int row = idx / (N4H / 8);
      const int c8  = idx - row * (N4H / 8);
      const _Float16* gp = xg + ((size_t)(b0 + row) * T + 0) * N4H + c8 * 8;
      unsigned lds = xgl_lds + (unsigned)idx * 16u;
      unsigned long long ga = (unsigned long long)(uintptr_t)gp;
      asm volatile("global_load_async_to_lds_b128 %0, %1, off"
                   :: "v"(lds), "v"(ga) : "memory");
    }
    asm volatile("s_wait_asynccnt 0x0" ::: "memory");
  }
  __syncthreads();

  float creg[TPW][8];
  #pragma unroll
  for (int tt = 0; tt < TPW; ++tt)
    #pragma unroll
    for (int r = 0; r < 8; ++r) creg[tt][r] = 0.0f;

  for (int t = 0; t < T; ++t) {
    // ---- init accumulators with xg tile from LDS (C layout)
    v8f acc[TPW];
    #pragma unroll
    for (int tt = 0; tt < TPW; ++tt) {
      const int tn = wave + tt * 8;
      if (tn < NT) {
        const int n0 = tn * 16;
        #pragma unroll
        for (int r = 0; r < 8; ++r) {
          const int m = mhi * 8 + r;
          acc[tt][r] = (float)xgl[m * N4H + n0 + mlo];
        }
      }
    }
    __syncthreads();   // everyone done reading xgl; safe to overwrite

    // ---- kick off async DMA of next step's tile, overlapped with WMMA
    if (t + 1 < T) {
      const int NOPS = 2 * N4H;
      for (int idx = tid; idx < NOPS; idx += 256) {
        const int row = idx / (N4H / 8);
        const int c8  = idx - row * (N4H / 8);
        const _Float16* gp = xg + ((size_t)(b0 + row) * T + (t + 1)) * N4H + c8 * 8;
        unsigned lds = xgl_lds + (unsigned)idx * 16u;
        unsigned long long ga = (unsigned long long)(uintptr_t)gp;
        asm volatile("global_load_async_to_lds_b128 %0, %1, off"
                     :: "v"(lds), "v"(ga) : "memory");
      }
    }

    // ---- g += h @ w_hh'^T via WMMA
    for (int k0 = 0; k0 < KP; k0 += 32) {
      v16h a;
      const _Float16* pa = hl + (size_t)mlo * KP + k0 + mhi * 8;
      *(v8h*)&a       = *(const v8h*)pa;
      *((v8h*)&a + 1) = *(const v8h*)(pa + 16);
      #pragma unroll
      for (int tt = 0; tt < TPW; ++tt) {
        const int tn = wave + tt * 8;
        if (tn < NT) {
          v16h bf;
          const _Float16* pb = Wl + (size_t)(tn * 16 + mlo) * KP + k0 + mhi * 8;
          *(v8h*)&bf       = *(const v8h*)pb;
          *((v8h*)&bf + 1) = *(const v8h*)(pb + 16);
          acc[tt] = __builtin_amdgcn_wmma_f32_16x16x32_f16(false, a, false, bf,
                                                           (short)0, acc[tt], false, false);
        }
      }
    }
    __syncthreads();     // all reads of hl done before rewrite

    // ---- gates: lanes 4j..4j+3 hold i,f,g,o of unit j -> combine via shuffles
    #pragma unroll
    for (int tt = 0; tt < TPW; ++tt) {
      const int tn = wave + tt * 8;           // wave-uniform guard
      if (tn >= NT) continue;
      const int n0 = tn * 16;
      #pragma unroll
      for (int r = 0; r < 8; ++r) {
        float s  = acc[tt][r];
        float s1 = __shfl_xor(s, 1, 32);
        float s2 = __shfl_xor(s, 2, 32);
        float s3 = __shfl_xor(s, 3, 32);
        if (gate == 0) {
          float ig = sigf(s);
          float fg = sigf(s1);
          float gg = tanhf(s2);
          float og = sigf(s3);
          float c  = fg * creg[tt][r] + ig * gg;
          creg[tt][r] = c;
          float h = og * tanhf(c);
          hl[(size_t)(mhi * 8 + r) * KP + ((n0 + mlo) >> 2)] = (_Float16)h;
        }
      }
    }
    __syncthreads();     // new h visible

    // ---- fused LayerNorm over H, write f16 sequence output
    {
      const int row = tid >> 4;          // 16 rows, 16 lanes each
      const int j0  = tid & 15;
      float sum = 0.0f, sq = 0.0f;
      for (int j = j0; j < H; j += 16) {
        float v = (float)hl[(size_t)row * KP + j];
        sum += v; sq += v * v;
      }
      #pragma unroll
      for (int m = 8; m >= 1; m >>= 1) {
        sum += __shfl_xor(sum, m, 32);
        sq  += __shfl_xor(sq,  m, 32);
      }
      const float mu  = sum * (1.0f / H);
      const float var = sq * (1.0f / H) - mu * mu;
      const float rs  = rsqrtf(var + 1e-5f);
      for (int j = j0; j < H; j += 16) {
        float v = (float)hl[(size_t)row * KP + j];
        seq_out[((size_t)(b0 + row) * T + t) * H + j] =
            (_Float16)((v - mu) * rs * gamma[j] + beta[j]);
      }
    }

    // ---- ensure next tile fully landed (own ops + all other waves via barrier)
    asm volatile("s_wait_asynccnt 0x0" ::: "memory");
    __syncthreads();
  }
}

// ---------------------------------------------------------------- MLP head
__global__ void head_mlp(const _Float16* __restrict__ seq3, int T,
                         const float* __restrict__ hw1, const float* __restrict__ hb1,
                         const float* __restrict__ hw2, const float* __restrict__ hb2,
                         const float* __restrict__ hw3, const float* __restrict__ hb3,
                         float* __restrict__ out, int B)
{
  const int b = blockIdx.x * blockDim.x + threadIdx.x;
  if (b >= B) return;
  float xv[48];
  const _Float16* p = seq3 + ((size_t)b * T + (T - 1)) * 48;
  #pragma unroll
  for (int j = 0; j < 48; ++j) xv[j] = (float)p[j];
  float z1[32];
  #pragma unroll 4
  for (int o = 0; o < 32; ++o) {
    float s = hb1[o];
    for (int j = 0; j < 48; ++j) s += hw1[o * 48 + j] * xv[j];
    z1[o] = 0.5f * s * (1.0f + erff(s * 0.7071067811865475f));
  }
  float z2[16];
  #pragma unroll 4
  for (int o = 0; o < 16; ++o) {
    float s = hb2[o];
    for (int j = 0; j < 32; ++j) s += hw2[o * 32 + j] * z1[j];
    z2[o] = 0.5f * s * (1.0f + erff(s * 0.7071067811865475f));
  }
  float s = hb3[0];
  #pragma unroll
  for (int j = 0; j < 16; ++j) s += hw3[j] * z2[j];
  out[b] = s;
}

// ---------------------------------------------------------------- launch
extern "C" void kernel_launch(void* const* d_in, const int* in_sizes, int n_in,
                              void* d_out, int out_size, void* d_ws, size_t ws_size,
                              hipStream_t stream) {
  (void)in_sizes; (void)n_in; (void)out_size; (void)ws_size;
  const float* x     = (const float*)d_in[0];
  const float* w_ih1 = (const float*)d_in[1];
  const float* w_hh1 = (const float*)d_in[2];
  const float* b_ih1 = (const float*)d_in[3];
  const float* b_hh1 = (const float*)d_in[4];
  const float* g1    = (const float*)d_in[5];
  const float* be1   = (const float*)d_in[6];
  const float* w_ih2 = (const float*)d_in[7];
  const float* w_hh2 = (const float*)d_in[8];
  const float* b_ih2 = (const float*)d_in[9];
  const float* b_hh2 = (const float*)d_in[10];
  const float* g2    = (const float*)d_in[11];
  const float* be2   = (const float*)d_in[12];
  const float* w_ih3 = (const float*)d_in[13];
  const float* w_hh3 = (const float*)d_in[14];
  const float* b_ih3 = (const float*)d_in[15];
  const float* b_hh3 = (const float*)d_in[16];
  const float* g3    = (const float*)d_in[17];
  const float* be3   = (const float*)d_in[18];
  const float* hw1   = (const float*)d_in[19];
  const float* hb1   = (const float*)d_in[20];
  const float* hw2   = (const float*)d_in[21];
  const float* hb2   = (const float*)d_in[22];
  const float* hw3   = (const float*)d_in[23];
  const float* hb3   = (const float*)d_in[24];

  const size_t BT = (size_t)Bb * Tt;
  _Float16* xh   = (_Float16*)d_ws;          // BT*64
  _Float16* seq1 = xh   + BT * 64;           // BT*192
  _Float16* seq2 = seq1 + BT * 192;          // BT*96
  _Float16* seq3 = seq2 + BT * 96;           // BT*48
  _Float16* xgb  = seq3 + BT * 48;           // BT*768 (shared by all layers)

  auto gk1 = gate_gemm<64, 768>;
  auto gk2 = gate_gemm<192, 384>;
  auto gk3 = gate_gemm<96, 192>;
  auto sk1 = lstm_scan<192, 768, 6>;
  auto sk2 = lstm_scan<96, 384, 3>;
  auto sk3 = lstm_scan<48, 192, 2>;

  const int smG1 = 768 * 64 * 2 + 768 * 4;                    // 101376
  const int smG2 = 384 * 192 * 2 + 384 * 4;                   // 148992
  const int smG3 = 192 * 96 * 2 + 192 * 4;                    // 37632
  const int smS1 = (768 * 192 + 16 * 192 + 16 * 768) * 2;     // 325632 (<= 320KB LDS)
  const int smS2 = (384 * 96 + 16 * 96 + 16 * 384) * 2;       // 89088
  const int smS3 = (192 * 64 + 16 * 64 + 16 * 192) * 2;       // 32768

  hipFuncSetAttribute(reinterpret_cast<const void*>(gk1), hipFuncAttributeMaxDynamicSharedMemorySize, smG1);
  hipFuncSetAttribute(reinterpret_cast<const void*>(gk2), hipFuncAttributeMaxDynamicSharedMemorySize, smG2);
  hipFuncSetAttribute(reinterpret_cast<const void*>(gk3), hipFuncAttributeMaxDynamicSharedMemorySize, smG3);
  hipFuncSetAttribute(reinterpret_cast<const void*>(sk1), hipFuncAttributeMaxDynamicSharedMemorySize, smS1);
  hipFuncSetAttribute(reinterpret_cast<const void*>(sk2), hipFuncAttributeMaxDynamicSharedMemorySize, smS2);
  hipFuncSetAttribute(reinterpret_cast<const void*>(sk3), hipFuncAttributeMaxDynamicSharedMemorySize, smS3);

  cvt_f32_f16<<<1024, 256, 0, stream>>>(x, xh, (int)(BT * 64));

  gk1<<<(int)(BT / 128), 256, smG1, stream>>>(xh,   w_ih1, b_ih1, b_hh1, xgb, (int)BT);
  sk1<<<Bb / 16, 256, smS1, stream>>>(xgb, w_hh1, g1, be1, seq1, Tt);

  gk2<<<(int)(BT / 128), 256, smG2, stream>>>(seq1, w_ih2, b_ih2, b_hh2, xgb, (int)BT);
  sk2<<<Bb / 16, 256, smS2, stream>>>(xgb, w_hh2, g2, be2, seq2, Tt);

  gk3<<<(int)(BT / 128), 256, smG3, stream>>>(seq2, w_ih3, b_ih3, b_hh3, xgb, (int)BT);
  sk3<<<Bb / 16, 256, smS3, stream>>>(xgb, w_hh3, g3, be3, seq3, Tt);

  head_mlp<<<(Bb + 255) / 256, 256, 0, stream>>>(seq3, Tt, hw1, hb1, hw2, hb2, hw3, hb3,
                                                 (float*)d_out, Bb);
}
